// Mamba3B_56942676411234
// MI455X (gfx1250) — matrix operand order repaired and checked by reading, not compiled, not run on hardware
//
#include <hip/hip_runtime.h>
#include <hip/hip_bf16.h>
#include <math.h>

// ---------------- model constants ----------------
#define N_LAYERS   2
#define D_MODEL    2560
#define D_INNER    5120
#define D_STATE    16
#define D_CONV     4
#define DT_RANK    160
#define NUM_LABELS 3
#define BB         2
#define LL         512
#define NTOK       (BB * LL)            // 1024 token rows
#define XDBL_COLS  (DT_RANK + 2 * D_STATE)   // 192

// ---------------- CDNA5 WMMA types ----------------
typedef __attribute__((ext_vector_type(16))) __bf16 v16bf;
typedef __attribute__((ext_vector_type(8)))  float  v8f;

union BfFrag { v16bf v; unsigned w[8]; };

// fp32x2 -> packed bf16x2: round-half-up then one v_perm_b32 grabs the two
// high halfwords (3 VALU per pair).
__device__ __forceinline__ unsigned pack_bf16x2(float lo, float hi) {
  unsigned a = __builtin_bit_cast(unsigned, lo) + 0x8000u;
  unsigned b = __builtin_bit_cast(unsigned, hi) + 0x8000u;
  // selector bytes [3..0] = {7,6,3,2}: dst = { b[31:16], a[31:16] }
  return __builtin_amdgcn_perm(b, a, 0x07060302u);
}

// A-side (fp32 on the fly): one 16-float contiguous run (K 0..15)
__device__ __forceinline__ void loadA(BfFrag& f, const float* p) {
  const float4* q = (const float4*)p;
  float4 a0 = q[0], a1 = q[1], a2 = q[2], a3 = q[3];
  f.w[0] = pack_bf16x2(a0.x, a0.y); f.w[1] = pack_bf16x2(a0.z, a0.w);
  f.w[2] = pack_bf16x2(a1.x, a1.y); f.w[3] = pack_bf16x2(a1.z, a1.w);
  f.w[4] = pack_bf16x2(a2.x, a2.y); f.w[5] = pack_bf16x2(a2.z, a2.w);
  f.w[6] = pack_bf16x2(a3.x, a3.y); f.w[7] = pack_bf16x2(a3.z, a3.w);
}

// B-side: 16 contiguous pre-converted bf16 -> two aligned uint4 loads, 0 VALU
__device__ __forceinline__ void loadBbf(BfFrag& f, const unsigned short* p) {
  const uint4* q = (const uint4*)p;
  uint4 x0 = q[0], x1 = q[1];
  f.w[0] = x0.x; f.w[1] = x0.y; f.w[2] = x0.z; f.w[3] = x0.w;
  f.w[4] = x1.x; f.w[5] = x1.y; f.w[6] = x1.z; f.w[7] = x1.w;
}

// =====================================================================
// Weight pre-conversion: fp32 -> bf16, 8 elements per thread (exact; all
// weight sizes are multiples of 2048). Pure bandwidth pass.
// =====================================================================
__global__ __launch_bounds__(256) void w_to_bf16_kernel(
    const float* __restrict__ src, unsigned* __restrict__ dst)
{
  const size_t i = (size_t)blockIdx.x * 256 + threadIdx.x;   // 8 floats each
  const float4* s = (const float4*)src;
  float4 a = s[2 * i], b = s[2 * i + 1];
  uint4 r;
  r.x = pack_bf16x2(a.x, a.y); r.y = pack_bf16x2(a.z, a.w);
  r.z = pack_bf16x2(b.x, b.y); r.w = pack_bf16x2(b.z, b.w);
  ((uint4*)dst)[i] = r;
}

// =====================================================================
// Generic GEMM: C[M,N] = act( A[M,K](f32) * Wbf[N,K]^T(bf16) + bias ) + resid
// bf16 WMMA, fp32 accumulate. Wave = 16(M) x 64(N): one A fragment feeds
// FOUR independent WMMA chains; B fragments are pure b128 loads (no VALU).
// REQUIRES N % 64 == 0 (true for all call sites).
// grid = ( ceil((N/64)/8), M/16 ), block = 256 (8 waves).
// act: 0 = none, 1 = softplus
// =====================================================================
__global__ __launch_bounds__(256) void gemm_wmma_bf16(
    const float* __restrict__ A, int lda,
    const unsigned short* __restrict__ W, int ldw,   // bf16 (N,K) row-major
    const float* __restrict__ bias,
    const float* __restrict__ resid,
    float* __restrict__ C, int ldc,
    int N, int K, int act)
{
  const int wave   = threadIdx.x >> 5;
  const int lane   = threadIdx.x & 31;
  const int hlf    = lane >> 4;        // 0: lanes 0-15, 1: lanes 16-31
  const int lane16 = lane & 15;

  const int mrow0 = blockIdx.y * 16;
  const int quad  = blockIdx.x * 8 + wave;       // 64-wide N strip index
  if (quad * 64 >= N) return;                    // wave-uniform idle exit

  v8f c0 = {}, c1 = {}, c2 = {}, c3 = {};

  // A: lane16 = row; half 0 covers K 0..15, half 1 covers K 16..31
  const float* aptr = A + (size_t)(mrow0 + lane16) * lda + (hlf ? 16 : 0);
  // B: lane16 = column; half 0 covers K 0..15, half 1 covers K 16..31
  const int ncol = quad * 64 + lane16;
  const unsigned short* w0 = W + (size_t)(ncol +  0) * ldw + (hlf ? 16 : 0);
  const unsigned short* w1 = W + (size_t)(ncol + 16) * ldw + (hlf ? 16 : 0);
  const unsigned short* w2 = W + (size_t)(ncol + 32) * ldw + (hlf ? 16 : 0);
  const unsigned short* w3 = W + (size_t)(ncol + 48) * ldw + (hlf ? 16 : 0);

  for (int k0 = 0; k0 < K; k0 += 32) {
    BfFrag af;  loadA(af, aptr + k0);

    BfFrag bf0; loadBbf(bf0, w0 + k0);
    c0 = __builtin_amdgcn_wmma_f32_16x16x32_bf16(false, af.v, false, bf0.v,
                                                 (short)0, c0, false, false);
    BfFrag bf1; loadBbf(bf1, w1 + k0);
    c1 = __builtin_amdgcn_wmma_f32_16x16x32_bf16(false, af.v, false, bf1.v,
                                                 (short)0, c1, false, false);
    BfFrag bf2; loadBbf(bf2, w2 + k0);
    c2 = __builtin_amdgcn_wmma_f32_16x16x32_bf16(false, af.v, false, bf2.v,
                                                 (short)0, c2, false, false);
    BfFrag bf3; loadBbf(bf3, w3 + k0);
    c3 = __builtin_amdgcn_wmma_f32_16x16x32_bf16(false, af.v, false, bf3.v,
                                                 (short)0, c3, false, false);

    // gfx1250 prefetch of next K tiles
    __builtin_prefetch(aptr + k0 + 64, 0, 3);
    __builtin_prefetch(w0 + k0 + 64, 0, 3);
    __builtin_prefetch(w2 + k0 + 64, 0, 3);
  }

  // ---- store: C VGPR j -> row (j + 8*half), col group g -> +16*g ----
  const int crow0 = mrow0 + (hlf << 3);
  v8f acc[4] = {c0, c1, c2, c3};
  #pragma unroll
  for (int g = 0; g < 4; ++g) {
    const int col = quad * 64 + g * 16 + lane16;
    float badd = bias ? bias[col] : 0.f;
    #pragma unroll
    for (int j = 0; j < 8; ++j) {
      const int row = crow0 + j;
      float v = acc[g][j] + badd;
      if (act == 1) v = (v > 20.f) ? v : log1pf(__expf(v));     // softplus
      if (resid) v += resid[(size_t)row * ldc + col];
      C[(size_t)row * ldc + col] = v;
    }
  }
}

// =====================================================================
// Embedding gather: one block per token row.
// =====================================================================
__global__ __launch_bounds__(256) void embed_kernel(
    const float* __restrict__ emb, const int* __restrict__ ids,
    float* __restrict__ x)
{
  const int row = blockIdx.x;
  const int id  = ids[row];
  const float* src = emb + (size_t)id * D_MODEL;
  float* dst = x + (size_t)row * D_MODEL;
  for (int i = threadIdx.x; i < D_MODEL; i += 256) dst[i] = src[i];
}

// =====================================================================
// RMSNorm: one block per token row (D = 2560).
// =====================================================================
__global__ __launch_bounds__(256) void rmsnorm_kernel(
    const float* __restrict__ x, const float* __restrict__ w,
    float* __restrict__ out, int D)
{
  __shared__ float red[256];
  const int tid = threadIdx.x;
  const size_t base = (size_t)blockIdx.x * D;
  float ss = 0.f;
  for (int i = tid; i < D; i += 256) { float v = x[base + i]; ss += v * v; }
  red[tid] = ss;
  __syncthreads();
  for (int s = 128; s > 0; s >>= 1) {
    if (tid < s) red[tid] += red[tid + s];
    __syncthreads();
  }
  const float scale = rsqrtf(red[0] / (float)D + 1e-5f);
  for (int i = tid; i < D; i += 256) out[base + i] = x[base + i] * scale * w[i];
}

// =====================================================================
// Depthwise causal conv1d (K=4) + bias + SiLU. xin = xz[:, 0:D_INNER].
// =====================================================================
__global__ __launch_bounds__(256) void conv_silu_kernel(
    const float* __restrict__ xz, const float* __restrict__ cw,
    const float* __restrict__ cb, float* __restrict__ xc)
{
  const size_t idx = (size_t)blockIdx.x * 256 + threadIdx.x;
  const int d   = (int)(idx % D_INNER);
  const int tok = (int)(idx / D_INNER);
  const int t   = tok & (LL - 1);
  const float* w = cw + (size_t)d * D_CONV;
  float acc = cb[d];
  #pragma unroll
  for (int k = 0; k < D_CONV; ++k) {
    const int tt = t - (D_CONV - 1) + k;
    if (tt >= 0)
      acc += w[k] * xz[(size_t)(tok - (D_CONV - 1) + k) * (2 * D_INNER) + d];
  }
  xc[(size_t)tok * D_INNER + d] = acc / (1.f + __expf(-acc));   // SiLU
}

// =====================================================================
// Selective scan: one thread per (batch, channel); 16-state recurrence in
// registers, B_t/C_t staged through LDS in 64-step chunks. y written in
// place over u (xc).
// =====================================================================
#define SCAN_CHUNK 64
__global__ __launch_bounds__(256) void scan_kernel(
    const float* __restrict__ delta, const float* __restrict__ xdbl,
    const float* __restrict__ A_log, const float* __restrict__ Dskip,
    float* __restrict__ xc)
{
  __shared__ float sB[SCAN_CHUNK][D_STATE];
  __shared__ float sC[SCAN_CHUNK][D_STATE];

  const int gid = blockIdx.x * 256 + threadIdx.x;
  const int d = gid % D_INNER;
  const int b = gid / D_INNER;           // uniform per block (5120 % 256 == 0)

  float a[D_STATE], h[D_STATE];
  #pragma unroll
  for (int n = 0; n < D_STATE; ++n) {
    a[n] = -__expf(A_log[(size_t)d * D_STATE + n]);
    h[n] = 0.f;
  }
  const float Dd = Dskip[d];

  for (int t0 = 0; t0 < LL; t0 += SCAN_CHUNK) {
    __syncthreads();
    for (int i = threadIdx.x; i < SCAN_CHUNK * 2 * D_STATE; i += 256) {
      const int tt = i >> 5;
      const int n  = i & 31;
      const float v = xdbl[(size_t)(b * LL + t0 + tt) * XDBL_COLS + DT_RANK + n];
      if (n < D_STATE) sB[tt][n] = v; else sC[tt][n - D_STATE] = v;
    }
    __syncthreads();
    for (int tt = 0; tt < SCAN_CHUNK; ++tt) {
      const size_t off = (size_t)(b * LL + t0 + tt) * D_INNER + d;
      const float u  = xc[off];
      const float dt = delta[off];
      const float du = dt * u;
      float y = 0.f;
      #pragma unroll
      for (int n = 0; n < D_STATE; ++n) {
        const float dA = __expf(dt * a[n]);
        h[n] = dA * h[n] + du * sB[tt][n];
        y += h[n] * sC[tt][n];
      }
      xc[off] = y + u * Dd;
    }
  }
}

// =====================================================================
// Gating: yg = y * silu(z), z = xz[:, D_INNER:2*D_INNER].
// =====================================================================
__global__ __launch_bounds__(256) void gate_kernel(
    const float* __restrict__ y, const float* __restrict__ xz,
    float* __restrict__ yg)
{
  const size_t idx = (size_t)blockIdx.x * 256 + threadIdx.x;
  const int d   = (int)(idx % D_INNER);
  const int tok = (int)(idx / D_INNER);
  const float z = xz[(size_t)tok * (2 * D_INNER) + D_INNER + d];
  yg[(size_t)tok * D_INNER + d] =
      y[(size_t)tok * D_INNER + d] * (z / (1.f + __expf(-z)));
}

// =====================================================================
// Masked mean-pool + classifier head. Single block (tiny).
// =====================================================================
__global__ __launch_bounds__(256) void pool_fc_kernel(
    const float* __restrict__ h, const int* __restrict__ mask,
    const float* __restrict__ fcw, const float* __restrict__ fcb,
    float* __restrict__ out)
{
  __shared__ float pooled[D_MODEL];
  __shared__ float red[256];
  const int tid = threadIdx.x;
  for (int b = 0; b < BB; ++b) {
    float cnt = 0.f;
    for (int t = 0; t < LL; ++t) cnt += (float)mask[b * LL + t];
    for (int i = tid; i < D_MODEL; i += 256) {
      float s = 0.f;
      for (int t = 0; t < LL; ++t)
        if (mask[b * LL + t]) s += h[(size_t)(b * LL + t) * D_MODEL + i];
      pooled[i] = s / cnt;
    }
    __syncthreads();
    for (int c = 0; c < NUM_LABELS; ++c) {
      float p = 0.f;
      for (int i = tid; i < D_MODEL; i += 256)
        p += pooled[i] * fcw[(size_t)c * D_MODEL + i];
      red[tid] = p;
      __syncthreads();
      for (int s = 128; s > 0; s >>= 1) {
        if (tid < s) red[tid] += red[tid + s];
        __syncthreads();
      }
      if (tid == 0) out[b * NUM_LABELS + c] = red[0] + fcb[c];
      __syncthreads();
    }
    __syncthreads();
  }
}

// =====================================================================
// Host orchestration
// =====================================================================
extern "C" void kernel_launch(void* const* d_in, const int* in_sizes, int n_in,
                              void* d_out, int out_size, void* d_ws, size_t ws_size,
                              hipStream_t stream) {
  (void)in_sizes; (void)n_in; (void)out_size; (void)ws_size;
  const float* emb        = (const float*)d_in[0];
  const float* norm_w     = (const float*)d_in[1];
  const float* in_proj_w  = (const float*)d_in[2];
  const float* conv_w     = (const float*)d_in[3];
  const float* conv_b     = (const float*)d_in[4];
  const float* x_proj_w   = (const float*)d_in[5];
  const float* dt_proj_w  = (const float*)d_in[6];
  const float* dt_proj_b  = (const float*)d_in[7];
  const float* A_log      = (const float*)d_in[8];
  const float* D_skip     = (const float*)d_in[9];
  const float* out_proj_w = (const float*)d_in[10];
  const float* norm_f_w   = (const float*)d_in[11];
  const float* fc_w       = (const float*)d_in[12];
  const float* fc_b       = (const float*)d_in[13];
  const int*   input_ids  = (const int*)d_in[14];
  const int*   attn_mask  = (const int*)d_in[15];
  float* out = (float*)d_out;

  // workspace carve-up (~158 MB)
  float* ws = (float*)d_ws;
  size_t o = 0;
  float* x     = ws + o; o += (size_t)NTOK * D_MODEL;       // residual stream
  float* hbuf  = ws + o; o += (size_t)NTOK * D_MODEL;       // rmsnorm out
  float* xz    = ws + o; o += (size_t)NTOK * 2 * D_INNER;   // in_proj out
  float* xc    = ws + o; o += (size_t)NTOK * D_INNER;       // conv out / scan y
  float* xdbl  = ws + o; o += (size_t)NTOK * XDBL_COLS;     // x_proj out
  float* delta = ws + o; o += (size_t)NTOK * D_INNER;       // dt / gated y
  unsigned* wbf = (unsigned*)(ws + o);                      // bf16 weight stage
  o += (size_t)2 * D_INNER * D_MODEL / 2;                   // 52.4 MB max

  const int mg = NTOK / 16;                                 // 64 M-tiles
  const int elemBlocks = (int)(((size_t)NTOK * D_INNER) / 256);
  #define GEMM_GX(N) (((N) / 64 + 7) / 8)
  #define CONV_W(src, nelem) \
    w_to_bf16_kernel<<<(int)((size_t)(nelem) / 2048), 256, 0, stream>>>((src), wbf)

  embed_kernel<<<NTOK, 256, 0, stream>>>(emb, input_ids, x);

  for (int i = 0; i < N_LAYERS; ++i) {
    rmsnorm_kernel<<<NTOK, 256, 0, stream>>>(x, norm_w + (size_t)i * D_MODEL,
                                             hbuf, D_MODEL);
    // xz = h @ in_proj_w^T  (N=10240, K=2560)
    CONV_W(in_proj_w + (size_t)i * 2 * D_INNER * D_MODEL,
           (size_t)2 * D_INNER * D_MODEL);
    gemm_wmma_bf16<<<dim3(GEMM_GX(2 * D_INNER), mg), 256, 0, stream>>>(
        hbuf, D_MODEL, (const unsigned short*)wbf, D_MODEL,
        nullptr, nullptr, xz, 2 * D_INNER, 2 * D_INNER, D_MODEL, 0);

    conv_silu_kernel<<<elemBlocks, 256, 0, stream>>>(
        xz, conv_w + (size_t)i * D_INNER * D_CONV,
        conv_b + (size_t)i * D_INNER, xc);

    // x_dbl = xc @ x_proj_w^T  (N=192, K=5120)
    CONV_W(x_proj_w + (size_t)i * XDBL_COLS * D_INNER,
           (size_t)XDBL_COLS * D_INNER);
    gemm_wmma_bf16<<<dim3(GEMM_GX(XDBL_COLS), mg), 256, 0, stream>>>(
        xc, D_INNER, (const unsigned short*)wbf, D_INNER,
        nullptr, nullptr, xdbl, XDBL_COLS, XDBL_COLS, D_INNER, 0);

    // delta = softplus(dt @ dt_proj_w^T + b)  (N=5120, K=160)
    CONV_W(dt_proj_w + (size_t)i * D_INNER * DT_RANK,
           (size_t)D_INNER * DT_RANK);
    gemm_wmma_bf16<<<dim3(GEMM_GX(D_INNER), mg), 256, 0, stream>>>(
        xdbl, XDBL_COLS, (const unsigned short*)wbf, DT_RANK,
        dt_proj_b + (size_t)i * D_INNER, nullptr, delta, D_INNER,
        D_INNER, DT_RANK, 1);

    scan_kernel<<<(BB * D_INNER) / 256, 256, 0, stream>>>(
        delta, xdbl, A_log + (size_t)i * D_INNER * D_STATE,
        D_skip + (size_t)i * D_INNER, xc);

    gate_kernel<<<elemBlocks, 256, 0, stream>>>(xc, xz, delta);

    // x += yg @ out_proj_w^T  (N=2560, K=5120), residual in place
    CONV_W(out_proj_w + (size_t)i * D_MODEL * D_INNER,
           (size_t)D_MODEL * D_INNER);
    gemm_wmma_bf16<<<dim3(GEMM_GX(D_MODEL), mg), 256, 0, stream>>>(
        delta, D_INNER, (const unsigned short*)wbf, D_INNER,
        nullptr, x, x, D_MODEL, D_MODEL, D_INNER, 0);
  }

  rmsnorm_kernel<<<NTOK, 256, 0, stream>>>(x, norm_f_w, hbuf, D_MODEL);
  pool_fc_kernel<<<1, 256, 0, stream>>>(hbuf, attn_mask, fc_w, fc_b, out);
}
